// PatchAttention_60593398612617
// MI455X (gfx1250) — matrix operand order repaired and checked by reading, not compiled, not run on hardware
//
#include <hip/hip_runtime.h>
#include <hip/hip_bf16.h>

// ---------------------------------------------------------------------------
// Types for CDNA5 WMMA (wave32, v_wmma_f32_16x16x32_bf16)
// ---------------------------------------------------------------------------
typedef __bf16 bf16_t;
typedef bf16_t v16bf __attribute__((ext_vector_type(16)));
typedef float  v8f   __attribute__((ext_vector_type(8)));

union Frag { v16bf v; unsigned int u[8]; };

__device__ __forceinline__ unsigned short f2bf(float f) {
  unsigned int u = __float_as_uint(f);
  u += 0x7fffu + ((u >> 16) & 1u);   // round-to-nearest-even
  return (unsigned short)(u >> 16);
}

// LDS byte offset of a __shared__ object (for async-to-LDS addressing)
__device__ __forceinline__ unsigned to_lds(const void* p) {
  return (unsigned)(unsigned long long)(__attribute__((address_space(3))) const void*)p;
}

// Async 16B copy global -> LDS (gfx1250 GLOBAL_LOAD_ASYNC_TO_LDS_B128, ASYNCcnt)
__device__ __forceinline__ void async_copy16(unsigned lds_off, const void* base,
                                             unsigned byte_off) {
  asm volatile("global_load_async_to_lds_b128 %0, %1, %2"
               :: "v"(lds_off), "v"(byte_off), "s"(base)
               : "memory");
}
template <int N>
__device__ __forceinline__ void async_waitN() {
  asm volatile("s_wait_asynccnt %0" :: "n"(N) : "memory");
}
__device__ __forceinline__ void async_wait0() {
  asm volatile("s_wait_asynccnt 0" ::: "memory");
}

// DPP16 ROW_XMASK butterfly (pure VALU cross-lane within 16-lane rows)
template <int CTRL>
__device__ __forceinline__ float dpp_xor_f(float x) {
  return __int_as_float(
      __builtin_amdgcn_update_dpp(0, __float_as_int(x), CTRL, 0xf, 0xf, true));
}
__device__ __forceinline__ float rowmax16(float x) {
  x = fmaxf(x, dpp_xor_f<0x161>(x));   // ROW_XMASK:1
  x = fmaxf(x, dpp_xor_f<0x162>(x));   // ROW_XMASK:2
  x = fmaxf(x, dpp_xor_f<0x164>(x));   // ROW_XMASK:4
  x = fmaxf(x, dpp_xor_f<0x168>(x));   // ROW_XMASK:8
  return x;
}
__device__ __forceinline__ float rowsum16(float x) {
  x += dpp_xor_f<0x161>(x);
  x += dpp_xor_f<0x162>(x);
  x += dpp_xor_f<0x164>(x);
  x += dpp_xor_f<0x168>(x);
  return x;
}

// ---------------------------------------------------------------------------
// Problem constants
// ---------------------------------------------------------------------------
#define BATCH 8
#define DIM   256
#define HP    32
#define NTOK  1024
#define MTOT  8192
#define HEADS 8
#define HD    32
#define TBL   3969
#define QSCALE 0.17677669529663687f

// ---------------------------------------------------------------------------
// Small prep kernels
// ---------------------------------------------------------------------------
__global__ void convert_f32_bf16(const float* __restrict__ in,
                                 unsigned short* __restrict__ out, int n) {
  int i = blockIdx.x * 256 + threadIdx.x;
  if (i < n) out[i] = f2bf(in[i]);
}

__global__ void transpose_wun(const float* __restrict__ w,
                              unsigned short* __restrict__ out) {
  int i = blockIdx.x * 256 + threadIdx.x;
  if (i >= 4096 * 256) return;
  int j = i >> 8;
  int c = i & 255;
  out[(size_t)j * 256 + c] = f2bf(w[(size_t)c * 4096 + j]);
}

__global__ void rpe_table(const float* __restrict__ w1, const float* __restrict__ b1,
                          const float* __restrict__ w2, float* __restrict__ tableT) {
  int i = blockIdx.x * 256 + threadIdx.x;
  if (i >= TBL) return;
  float rh = ((float)(i / 63) - 31.f) * (1.f / 31.f);
  float rw = ((float)(i % 63) - 31.f) * (1.f / 31.f);
  float acc[8] = {0, 0, 0, 0, 0, 0, 0, 0};
  for (int j = 0; j < 512; ++j) {
    float hv = fmaf(rh, w1[2 * j], fmaf(rw, w1[2 * j + 1], b1[j]));
    hv = fmaxf(hv, 0.f);
#pragma unroll
    for (int hh = 0; hh < 8; ++hh) acc[hh] = fmaf(hv, w2[hh * 512 + j], acc[hh]);
  }
#pragma unroll
  for (int hh = 0; hh < 8; ++hh) tableT[hh * TBL + i] = acc[hh];
}

// ---------------------------------------------------------------------------
// WMMA GEMM: out[M][N] = A[M][K] * Bt[N][K]^T, 128x128 block tile, K tile 64.
// Double-buffered LDS; tiles filled with GLOBAL_LOAD_ASYNC_TO_LDS_B128
// (A for modes 1-3, B for all modes) overlapped with the previous tile's
// WMMA work; mode 0 fuses im2col + f32->bf16 conversion for A.
// ---------------------------------------------------------------------------
template <int MODE>
__global__ __launch_bounds__(256) void gemm_bf16(
    const float* __restrict__ Xin,
    const unsigned short* __restrict__ Abf,
    const unsigned short* __restrict__ Bt,
    const float* __restrict__ bias,
    unsigned short* __restrict__ Obf,
    float* __restrict__ Of,
    unsigned short* __restrict__ Qo,
    unsigned short* __restrict__ Ko,
    unsigned short* __restrict__ Vto,
    int M, int N, int K) {
  __shared__ __align__(16) unsigned short As[2][128 * 72];  // 144B rows (16B aligned)
  __shared__ __align__(16) unsigned short Bs[2][128 * 72];

  const int tid = threadIdx.x;
  const int lane = tid & 31;
  const int wv = tid >> 5;
  const int h2 = lane >> 4;
  const int l16 = lane & 15;
  const int wave_m = wv >> 1;
  const int wave_n = wv & 1;
  const int m0 = blockIdx.y * 128;
  const int n0 = blockIdx.x * 128;

  const unsigned as_base = to_lds(As);
  const unsigned bs_base = to_lds(Bs);

  // issue all copies/fills for the K-tile starting at k0 into buffer `bufi`
  auto issue_tile = [&](int k0, int bufi) {
    const unsigned asb = as_base + (unsigned)(bufi * 128 * 144);
    const unsigned bsb = bs_base + (unsigned)(bufi * 128 * 144);
    // B tile: 128x64 bf16 = 1024 16B units, 4 async instructions per wave
#pragma unroll
    for (int i = 0; i < 4; ++i) {
      int u = tid + i * 256;
      int row = u >> 3, seg = u & 7;
      async_copy16(bsb + (unsigned)(row * 144 + seg * 16), Bt,
                   (unsigned)(((n0 + row) * K + k0 + seg * 8) * 2));
    }
    if (MODE == 0) {
      // fused im2col from x[B][C][128][128], convert f32->bf16 in VGPRs
      unsigned short* Asb = &As[bufi][0];
#pragma unroll
      for (int i = 0; i < 8; ++i) {
        int e = tid + i * 256;
        int row = e >> 4, c4 = e & 15;
        int kk = k0 + c4 * 4;
        int m = m0 + row;
        int c = kk >> 4, p = (kk >> 2) & 3;
        int b = m >> 10, tok = m & 1023, hp = tok >> 5, wp = tok & 31;
        const float4 xv =
            *(const float4*)(Xin + ((size_t)((b * 256 + c) * 128 + hp * 4 + p)) * 128 + wp * 4);
        unsigned lo = (unsigned)f2bf(xv.x) | ((unsigned)f2bf(xv.y) << 16);
        unsigned hi = (unsigned)f2bf(xv.z) | ((unsigned)f2bf(xv.w) << 16);
        *(uint2*)(Asb + row * 72 + c4 * 4) = make_uint2(lo, hi);
      }
    } else {
#pragma unroll
      for (int i = 0; i < 4; ++i) {
        int u = tid + i * 256;
        int row = u >> 3, seg = u & 7;
        async_copy16(asb + (unsigned)(row * 144 + seg * 16), Abf,
                     (unsigned)(((m0 + row) * K + k0 + seg * 8) * 2));
      }
    }
  };

  v8f zero = {};
  v8f acc[2][4];
#pragma unroll
  for (int sm = 0; sm < 2; ++sm)
#pragma unroll
    for (int sn = 0; sn < 4; ++sn) acc[sm][sn] = zero;

  const int nk = K >> 6;
  issue_tile(0, 0);

  for (int kt = 0; kt < nk; ++kt) {
    // prefetch next tile into the other buffer, then wait for this tile
    if (kt + 1 < nk) {
      issue_tile((kt + 1) << 6, (kt + 1) & 1);
      if (MODE == 0) async_waitN<4>(); else async_waitN<8>();
    } else {
      async_wait0();
    }
    __syncthreads();

    const unsigned short* Asb = &As[kt & 1][0];
    const unsigned short* Bsb = &Bs[kt & 1][0];
#pragma unroll
    for (int s = 0; s < 2; ++s) {
      Frag afr[2], bfr[4];
#pragma unroll
      for (int sm = 0; sm < 2; ++sm) {
        int Mrow = wave_m * 32 + sm * 16 + l16;
#pragma unroll
        for (int d = 0; d < 8; ++d) {
          int k = s * 32 + h2 * 16 + 2 * d;
          afr[sm].u[d] = *(const unsigned int*)(Asb + Mrow * 72 + k);
        }
      }
#pragma unroll
      for (int sn = 0; sn < 4; ++sn) {
        int nrow = wave_n * 64 + sn * 16 + l16;
        const uint4 t0 = *(const uint4*)(Bsb + nrow * 72 + s * 32 + h2 * 16);
        const uint4 t1 = *(const uint4*)(Bsb + nrow * 72 + s * 32 + h2 * 16 + 8);
        bfr[sn].u[0] = t0.x; bfr[sn].u[1] = t0.y; bfr[sn].u[2] = t0.z; bfr[sn].u[3] = t0.w;
        bfr[sn].u[4] = t1.x; bfr[sn].u[5] = t1.y; bfr[sn].u[6] = t1.z; bfr[sn].u[7] = t1.w;
      }
#pragma unroll
      for (int sm = 0; sm < 2; ++sm)
#pragma unroll
        for (int sn = 0; sn < 4; ++sn)
          acc[sm][sn] = __builtin_amdgcn_wmma_f32_16x16x32_bf16(
              false, afr[sm].v, false, bfr[sn].v, (short)0, acc[sm][sn], false, false);
    }
    __syncthreads();   // all waves done with buf[kt&1] before it is refilled
  }

  // ---- epilogue ----
#pragma unroll
  for (int sm = 0; sm < 2; ++sm) {
#pragma unroll
    for (int sn = 0; sn < 4; ++sn) {
      int n_g = n0 + wave_n * 64 + sn * 16 + l16;
#pragma unroll
      for (int v = 0; v < 8; ++v) {
        int m_g = m0 + wave_m * 32 + sm * 16 + v + 8 * h2;
        float val = acc[sm][sn][v];
        if (MODE == 0 || MODE == 2) {
          val += bias[n_g];
          Obf[(size_t)m_g * N + n_g] = f2bf(val);
        } else if (MODE == 1) {
          val += bias[n_g];
          int comp = n_g >> 8;
          int c = n_g & 255;
          int h = c >> 5, dh = c & 31;
          int b = m_g >> 10, t = m_g & 1023;
          size_t bh = (size_t)(b * 8 + h);
          if (comp == 0) {
            Qo[(bh * 1024 + t) * 32 + dh] = f2bf(val * QSCALE);
          } else if (comp == 1) {
            Ko[(bh * 1024 + t) * 32 + dh] = f2bf(val);
          } else {
            Vto[(bh * 32 + dh) * 1024 + t] = f2bf(val);
          }
        } else {  // MODE == 3: conv-transpose scatter, f32 + bias[o]
          int o = n_g >> 4, p = (n_g >> 2) & 3, q = n_g & 3;
          float vb = val + bias[o];
          int b = m_g >> 10, t = m_g & 1023, hp = t >> 5, wp = t & 31;
          Of[(((size_t)(b * 256 + o) * 128) + hp * 4 + p) * 128 + wp * 4 + q] = vb;
        }
      }
    }
  }
}

// ---------------------------------------------------------------------------
// Flash attention with dynamic RPE bias.
// Grid (64 = B*HEADS, 8 row slabs). Block 256 = 8 waves, 16 query rows/wave.
// K/V chunks double-buffered in LDS via async copies (waves 0-3 stage K,
// waves 4-7 stage V -> branch is wave-uniform); DPP-xmask online softmax;
// P restaged through per-wave LDS to form the WMMA A fragment.
// ---------------------------------------------------------------------------
__global__ __launch_bounds__(256) void attn_fwd(
    const unsigned short* __restrict__ Q,    // [bh][1024][32] bf16 (pre-scaled)
    const unsigned short* __restrict__ Kt,   // [bh][1024][32] bf16
    const unsigned short* __restrict__ Vt,   // [bh][32][1024] bf16
    const float* __restrict__ tableT,        // [HEADS][TBL] f32
    unsigned short* __restrict__ Oout) {     // [8192][256] bf16
  __shared__ float biasl[TBL];
  __shared__ __align__(16) unsigned short Klds[2][32 * 32];  // [tok][dh]
  __shared__ __align__(16) unsigned short Vlds[2][32 * 32];  // [dh][tok]
  __shared__ __align__(16) unsigned short Plds[8 * 16 * 32];

  const int tid = threadIdx.x;
  const int lane = tid & 31;
  const int wv = tid >> 5;
  const int h2 = lane >> 4;
  const int l16 = lane & 15;
  const int bh = blockIdx.x;
  const int b = bh >> 3;
  const int h = bh & 7;
  const int row_base = blockIdx.y * 128 + wv * 16;

  const unsigned k_base = to_lds(Klds);
  const unsigned v_base = to_lds(Vlds);

  // stage one 32-key K/V chunk into buffer `bufi` (1 async instr per wave)
  auto issue_kv = [&](int c0, int bufi) {
    if (tid < 128) {
      int row = tid >> 2, e = tid & 3;   // K: [32 tok][32 dh], 64B rows
      async_copy16(k_base + (unsigned)(bufi * 2048 + row * 64 + e * 16), Kt,
                   (unsigned)((((bh * 1024) + c0 + row) * 32 + e * 8) * 2));
    } else {
      int t2 = tid - 128;
      int row = t2 >> 2, e = t2 & 3;     // V: [32 dh][32 tok]
      async_copy16(v_base + (unsigned)(bufi * 2048 + row * 64 + e * 16), Vt,
                   (unsigned)((((bh * 32) + row) * 1024 + c0 + e * 8) * 2));
    }
  };

  for (int i = tid; i < TBL; i += 256) biasl[i] = tableT[h * TBL + i];

  // Q fragment (A-matrix 16x32)
  Frag qa;
  {
    const unsigned short* qp = Q + ((size_t)bh * 1024 + row_base + l16) * 32;
#pragma unroll
    for (int d = 0; d < 8; ++d) {
      int k = h2 * 16 + 2 * d;
      qa.u[d] = *(const unsigned int*)(qp + k);
    }
  }

  v8f zero = {};
  v8f o0 = zero, o1 = zero;
  float mrow[8], lrow[8];
#pragma unroll
  for (int v = 0; v < 8; ++v) { mrow[v] = -1e30f; lrow[v] = 0.f; }

  unsigned short* pw = Plds + wv * 512;

  issue_kv(0, 0);
  for (int ci = 0; ci < 32; ++ci) {
    const int c0 = ci * 32;
    if (ci + 1 < 32) {
      issue_kv(c0 + 32, (ci + 1) & 1);
      async_waitN<1>();
    } else {
      async_wait0();
    }
    __syncthreads();

    const unsigned short* Kb_ = &Klds[ci & 1][0];
    const unsigned short* Vb_ = &Vlds[ci & 1][0];

    // K fragments from LDS (B-matrix 32x16)
    Frag kb0, kb1;
    {
      const unsigned short* kp0 = Kb_ + l16 * 32 + h2 * 16;
      const unsigned short* kp1 = kp0 + 16 * 32;
      uint4 t0 = *(const uint4*)kp0;
      uint4 t1 = *(const uint4*)(kp0 + 8);
      kb0.u[0] = t0.x; kb0.u[1] = t0.y; kb0.u[2] = t0.z; kb0.u[3] = t0.w;
      kb0.u[4] = t1.x; kb0.u[5] = t1.y; kb0.u[6] = t1.z; kb0.u[7] = t1.w;
      uint4 t2 = *(const uint4*)kp1;
      uint4 t3 = *(const uint4*)(kp1 + 8);
      kb1.u[0] = t2.x; kb1.u[1] = t2.y; kb1.u[2] = t2.z; kb1.u[3] = t2.w;
      kb1.u[4] = t3.x; kb1.u[5] = t3.y; kb1.u[6] = t3.z; kb1.u[7] = t3.w;
    }
    v8f s0 = __builtin_amdgcn_wmma_f32_16x16x32_bf16(false, qa.v, false, kb0.v,
                                                     (short)0, zero, false, false);
    v8f s1 = __builtin_amdgcn_wmma_f32_16x16x32_bf16(false, qa.v, false, kb1.v,
                                                     (short)0, zero, false, false);

    // bias add + online softmax (row = v + 8*h2); DPP xmask reductions
#pragma unroll
    for (int v = 0; v < 8; ++v) {
      int r = row_base + v + 8 * h2;
      int rh = r >> 5, rw = r & 31;
      int cA = c0 + l16;
      int cB = c0 + 16 + l16;
      float sv0 = s0[v] + biasl[(rh - (cA >> 5) + 31) * 63 + (rw - (cA & 31) + 31)];
      float sv1 = s1[v] + biasl[(rh - (cB >> 5) + 31) * 63 + (rw - (cB & 31) + 31)];
      float mx = rowmax16(fmaxf(sv0, sv1));
      float mnew = fmaxf(mrow[v], mx);
      float corr = __expf(mrow[v] - mnew);
      float p0 = __expf(sv0 - mnew);
      float p1 = __expf(sv1 - mnew);
      float rs = rowsum16(p0 + p1);
      lrow[v] = lrow[v] * corr + rs;
      mrow[v] = mnew;
      o0[v] *= corr;
      o1[v] *= corr;
      int prow = v + 8 * h2;
      pw[prow * 32 + l16] = f2bf(p0);
      pw[prow * 32 + 16 + l16] = f2bf(p1);
    }
    asm volatile("s_wait_dscnt 0" ::: "memory");

    // P fragment (A-matrix 16x32) from per-wave LDS staging
    Frag pa;
#pragma unroll
    for (int d = 0; d < 8; ++d) {
      int k = h2 * 16 + 2 * d;
      pa.u[d] = *(const unsigned int*)(pw + l16 * 32 + k);
    }

    // V fragments from LDS (B-matrix 32x16)
    Frag vb0, vb1;
    {
      const unsigned short* vp0 = Vb_ + l16 * 32 + h2 * 16;
      const unsigned short* vp1 = vp0 + 16 * 32;
      uint4 t0 = *(const uint4*)vp0;
      uint4 t1 = *(const uint4*)(vp0 + 8);
      vb0.u[0] = t0.x; vb0.u[1] = t0.y; vb0.u[2] = t0.z; vb0.u[3] = t0.w;
      vb0.u[4] = t1.x; vb0.u[5] = t1.y; vb0.u[6] = t1.z; vb0.u[7] = t1.w;
      uint4 t2 = *(const uint4*)vp1;
      uint4 t3 = *(const uint4*)(vp1 + 8);
      vb1.u[0] = t2.x; vb1.u[1] = t2.y; vb1.u[2] = t2.z; vb1.u[3] = t2.w;
      vb1.u[4] = t3.x; vb1.u[5] = t3.y; vb1.u[6] = t3.z; vb1.u[7] = t3.w;
    }
    o0 = __builtin_amdgcn_wmma_f32_16x16x32_bf16(false, pa.v, false, vb0.v,
                                                 (short)0, o0, false, false);
    o1 = __builtin_amdgcn_wmma_f32_16x16x32_bf16(false, pa.v, false, vb1.v,
                                                 (short)0, o1, false, false);
    __syncthreads();   // all waves done with buf[ci&1] before it is refilled
  }

  // normalize and store
#pragma unroll
  for (int v = 0; v < 8; ++v) {
    float inv = 1.0f / lrow[v];
    int r = row_base + v + 8 * h2;
    size_t base = ((size_t)(b * 1024 + r)) * 256 + h * 32;
    Oout[base + l16] = f2bf(o0[v] * inv);
    Oout[base + 16 + l16] = f2bf(o1[v] * inv);
  }
}

// ---------------------------------------------------------------------------
// Launch
// ---------------------------------------------------------------------------
extern "C" void kernel_launch(void* const* d_in, const int* in_sizes, int n_in,
                              void* d_out, int out_size, void* d_ws, size_t ws_size,
                              hipStream_t stream) {
  const float* x         = (const float*)d_in[0];
  const float* w_embed   = (const float*)d_in[1];
  const float* b_embed   = (const float*)d_in[2];
  const float* cpb_w1    = (const float*)d_in[3];
  const float* cpb_b1    = (const float*)d_in[4];
  const float* cpb_w2    = (const float*)d_in[5];
  const float* w_qkv     = (const float*)d_in[6];
  const float* b_qkv     = (const float*)d_in[7];
  const float* w_proj    = (const float*)d_in[8];
  const float* b_proj    = (const float*)d_in[9];
  const float* w_unembed = (const float*)d_in[10];
  const float* b_unembed = (const float*)d_in[11];
  float* out = (float*)d_out;
  (void)in_sizes; (void)n_in; (void)out_size; (void)ws_size;

  char* base = (char*)d_ws;
  size_t off = 0;
  auto alloc = [&](size_t bytes) -> char* {
    char* p = base + off;
    off += (bytes + 255) & ~(size_t)255;
    return p;
  };
  unsigned short* tok     = (unsigned short*)alloc((size_t)MTOT * 256 * 2);
  unsigned short* Qb      = (unsigned short*)alloc((size_t)64 * 1024 * 32 * 2);
  unsigned short* Kb      = (unsigned short*)alloc((size_t)64 * 1024 * 32 * 2);
  unsigned short* Vtb     = (unsigned short*)alloc((size_t)64 * 32 * 1024 * 2);
  unsigned short* attn_o  = (unsigned short*)alloc((size_t)MTOT * 256 * 2);
  unsigned short* z2      = (unsigned short*)alloc((size_t)MTOT * 256 * 2);
  unsigned short* Wemb_t  = (unsigned short*)alloc((size_t)4096 * 256 * 2);
  unsigned short* Wqkv_t  = (unsigned short*)alloc((size_t)768 * 256 * 2);
  unsigned short* Wproj_t = (unsigned short*)alloc((size_t)256 * 256 * 2);
  unsigned short* Wun_t   = (unsigned short*)alloc((size_t)4096 * 256 * 2);
  float* tableT           = (float*)alloc((size_t)HEADS * TBL * 4);

  convert_f32_bf16<<<(1048576 + 255) / 256, 256, 0, stream>>>(w_embed, Wemb_t, 1048576);
  convert_f32_bf16<<<(196608 + 255) / 256, 256, 0, stream>>>(w_qkv, Wqkv_t, 196608);
  convert_f32_bf16<<<(65536 + 255) / 256, 256, 0, stream>>>(w_proj, Wproj_t, 65536);
  transpose_wun<<<(1048576 + 255) / 256, 256, 0, stream>>>(w_unembed, Wun_t);
  rpe_table<<<(TBL + 255) / 256, 256, 0, stream>>>(cpb_w1, cpb_b1, cpb_w2, tableT);

  // 1) patch embed (fused im2col): [8192 x 4096] x [4096 x 256]
  gemm_bf16<0><<<dim3(2, 64), 256, 0, stream>>>(x, nullptr, Wemb_t, b_embed, tok,
                                                nullptr, nullptr, nullptr, nullptr,
                                                MTOT, 256, 4096);
  // 2) qkv: [8192 x 256] x [256 x 768]
  gemm_bf16<1><<<dim3(6, 64), 256, 0, stream>>>(nullptr, tok, Wqkv_t, b_qkv, nullptr,
                                                nullptr, Qb, Kb, Vtb,
                                                MTOT, 768, 256);
  // 3) flash attention with RPE bias
  attn_fwd<<<dim3(64, 8), 256, 0, stream>>>(Qb, Kb, Vtb, tableT, attn_o);
  // 4) proj: [8192 x 256] x [256 x 256]
  gemm_bf16<2><<<dim3(2, 64), 256, 0, stream>>>(nullptr, attn_o, Wproj_t, b_proj, z2,
                                                nullptr, nullptr, nullptr, nullptr,
                                                MTOT, 256, 256);
  // 5) patch unembed: [8192 x 256] x [256 x 4096] -> scatter f32 output
  gemm_bf16<3><<<dim3(32, 64), 256, 0, stream>>>(nullptr, z2, Wun_t, b_unembed, nullptr,
                                                 out, nullptr, nullptr, nullptr,
                                                 MTOT, 4096, 256);
}
